// STP_UDGAT_9612136808805
// MI455X (gfx1250) — matrix operand order repaired and checked by reading, not compiled, not run on hardware
//
#include <hip/hip_runtime.h>
#include <stdint.h>

// ---------------------------------------------------------------------------
// STP-UDGAT forward for MI455X (gfx1250, wave32, WMMA + TDM).
// GEMMs: A panel (64 rows x K) DMA'd into LDS by the Tensor Data Mover
// (tensor_load_to_lds, TENSORcnt) with hardware row padding for bank-conflict
// free ds_load_b128 fragment reads; 4 waves/block each register-block 4x2
// 16x16 tiles -> 8 v_wmma_f32_16x16x32_bf16 per K-step, B streamed from the
// pre-transposed bf16 weights (L2-resident).
// Edge softmax: ordered-uint atomicMax + f32 atomicAdd scatter.
// ---------------------------------------------------------------------------

typedef __bf16 bf16;
typedef __attribute__((ext_vector_type(16))) __bf16 v16bf;
typedef __attribute__((ext_vector_type(8)))  float  v8f;

#define HC    256   // H*C
#define HEADS 4
#define CH    64
#define FIN   128
#define NEG_SLOPE 0.2f

#define MT 4            // 16-row tiles per wave  (64 rows)
#define NT 2            // 16-col tiles per wave  (32 cols)
#define BLK_ROWS 64
#define BLK_COLS 128    // 4 waves * 32 cols
#define LDS_PAD  8      // bf16 elements of pad per LDS row (16B, keeps b128 align)

#if defined(__HIP_DEVICE_COMPILE__) && __has_builtin(__builtin_amdgcn_tensor_load_to_lds)
#define HAS_TDM 1
#else
#define HAS_TDM 0
#endif

// ---- order-preserving float <-> uint key (for atomic segment-max) ----------
__device__ __forceinline__ unsigned fkey(float f) {
  unsigned u = __float_as_uint(f);
  return (u & 0x80000000u) ? ~u : (u | 0x80000000u);
}
__device__ __forceinline__ float fdecode(unsigned k) {
  unsigned u = (k & 0x80000000u) ? (k & 0x7FFFFFFFu) : ~k;
  return __uint_as_float(u);
}

#if HAS_TDM
typedef __attribute__((ext_vector_type(4))) unsigned tdm_v4u;
typedef __attribute__((ext_vector_type(8))) int      tdm_v8i;
typedef __attribute__((ext_vector_type(4))) int      tdm_v4i;

// 2-D bf16 tile load via Tensor Data Mover (ISA 08: D# groups 0/1, type=2).
// rows x rowElems tile, global row stride = strideElems, zero-fill past
// tensorRows (OOB rows), LDS row padding of 4 dwords via pad_enable.
__device__ __forceinline__ void tdm_load_2d(unsigned ldsOff, const void* gaddr,
                                            unsigned rows, unsigned rowElems,
                                            unsigned strideElems,
                                            unsigned tensorRows) {
  unsigned long long ga = (unsigned long long)(size_t)gaddr;
  tdm_v4u g0;
  g0[0] = 1u;                                   // count=1 (valid), user mode
  g0[1] = ldsOff;                               // lds_addr (bytes)
  g0[2] = (unsigned)(ga & 0xFFFFFFFFull);       // global_addr[31:0]
  g0[3] = (unsigned)((ga >> 32) & 0x01FFFFFFull) | (2u << 30);  // addr[56:32], type=2
  // pad_interval code: one row = rowElems/2 dwords = 2^(code+1)
  unsigned rowDw = rowElems >> 1;
  unsigned padI = 31u - (unsigned)__clz(rowDw) - 1u;
  unsigned padA = (LDS_PAD / 2) - 1;            // pad dwords - 1
  unsigned w0 = (1u << 16)                      // data_size = 2 bytes
              | (1u << 20)                      // pad_enable
              | (padI << 22) | (padA << 25);
  tdm_v8i g1;
  g1[0] = (int)w0;
  g1[1] = (int)((strideElems & 0xFFFFu) << 16);                 // tensor_dim0 lo16
  g1[2] = (int)((strideElems >> 16) | ((tensorRows & 0xFFFFu) << 16)); // dim0 hi / dim1 lo
  g1[3] = (int)((tensorRows >> 16) | (rowElems << 16));         // dim1 hi / tile_dim0
  g1[4] = (int)rows;                                            // tile_dim1 (dim2=0)
  g1[5] = (int)strideElems;                                     // tensor_dim0_stride lo32
  g1[6] = 0;                                                    // stride0 hi / stride1 lo
  g1[7] = 0;
  tdm_v4i z4 = {0, 0, 0, 0};
  tdm_v8i z8 = {0, 0, 0, 0, 0, 0, 0, 0};
  // 6-arg variant (clang-23 / therock-10.0 headers): groups 0-3 + extra + cpol
  __builtin_amdgcn_tensor_load_to_lds(g0, g1, z4, z4, z8, 0);
}
#endif

// ---- f32 -> bf16 transpose/convert: out[b][c][r] = in[b][r][c] -------------
__global__ void k_transpose_bf16(const float* __restrict__ in,
                                 bf16* __restrict__ out, int R, int C) {
  long long base = (long long)blockIdx.z * R * C;
  int t = blockIdx.x * blockDim.x + threadIdx.x;
  if (t >= R * C) return;
  int r = t / C, c = t - r * C;
  out[base + (long long)c * R + r] = (bf16)in[base + t];
}

// ---- gather rows + convert to bf16 -----------------------------------------
__global__ void k_gather_bf16(const float* __restrict__ emb,
                              const int* __restrict__ idx,
                              bf16* __restrict__ out, int N, int F) {
  int t = blockIdx.x * blockDim.x + threadIdx.x;
  if (t >= N * F) return;
  int n = t / F, f = t - n * F;
  int row = idx ? idx[n] : n;
  out[t] = (bf16)emb[(long long)row * F + f];
}

// ---- blocked bf16 WMMA GEMM: C[M,N] = A[M,K] * Bt[N,K]^T (+bias) -----------
__global__ void __launch_bounds__(128)
k_wmma_gemm(const bf16* __restrict__ A, int lda,
            const bf16* __restrict__ Bt, int ldbt,
            float* __restrict__ C, int ldc,
            bf16* __restrict__ Cbf, int ldcb,
            const float* __restrict__ bias,
            int M, int N, int K, int zA, int zB, int zC) {
  extern __shared__ bf16 sA[];              // [BLK_ROWS][K + LDS_PAD]
  A  += (long long)blockIdx.z * zA;
  Bt += (long long)blockIdx.z * zB;
  if (C)   C   += (long long)blockIdx.z * zC;
  if (Cbf) Cbf += (long long)blockIdx.z * zC;

  const int row0 = blockIdx.x * BLK_ROWS;
  const int colB = blockIdx.y * BLK_COLS;
  const int tid  = threadIdx.x;
  const int P = K + LDS_PAD;                // padded LDS row pitch (elements)

#if HAS_TDM
  // ---- A panel: one TDM descriptor, issued by wave 0; OOB rows zero-fill ---
  if (tid < 32) {
    tdm_load_2d((unsigned)(size_t)sA, A + (long long)row0 * lda,
                BLK_ROWS, (unsigned)K, (unsigned)lda, (unsigned)(M - row0));
    __builtin_amdgcn_s_wait_tensorcnt(0);
  }
  __syncthreads();
#else
  // ---- fallback: cooperative b128 fill ----
  const int chunksPerRow = K >> 3;          // uint4 = 8 bf16
  const int totalChunks = BLK_ROWS * chunksPerRow;
  for (int c = tid; c < totalChunks; c += blockDim.x) {
    int r = c / chunksPerRow, k8 = c - r * chunksPerRow;
    int row = row0 + r; if (row > M - 1) row = M - 1;
    *(uint4*)(sA + r * P + (k8 << 3)) =
        *(const uint4*)(A + (long long)row * lda + (k8 << 3));
  }
  __syncthreads();
#endif

  const int wave = tid >> 5, lane = tid & 31;
  const int col0 = colB + wave * (NT * 16);
  const int mi = lane & 15, kg = lane >> 4;

  // B row pointers per N-tile (clamped at the N edge; stores are guarded)
  const bf16* Brow[NT];
#pragma unroll
  for (int j = 0; j < NT; ++j) {
    int cb = col0 + (j << 4); if (cb > N - 16) cb = N - 16;
    Brow[j] = Bt + (long long)(cb + mi) * ldbt + (kg << 3);
  }

  union Frag { v16bf v; uint4 q[2]; };
  v8f acc[MT][NT];
#pragma unroll
  for (int i = 0; i < MT; ++i)
#pragma unroll
    for (int j = 0; j < NT; ++j) acc[i][j] = (v8f){};

  for (int kk = 0; kk < K; kk += 32) {
    Frag a[MT], b[NT];
    // A fragments from LDS: two b128 reads per fragment
    // (ISA 7.12.2: dwords 0-3 -> K = kk+kg*8 + {0,2,4,6}; dwords 4-7 -> +16)
#pragma unroll
    for (int i = 0; i < MT; ++i) {
      const bf16* ap = sA + (size_t)((i << 4) + mi) * P + (kg << 3) + kk;
      a[i].q[0] = *(const uint4*)(ap);
      a[i].q[1] = *(const uint4*)(ap + 16);
    }
    // B fragments streamed from global (pre-transposed; L2 resident)
#pragma unroll
    for (int j = 0; j < NT; ++j) {
      const bf16* bp = Brow[j] + kk;
      b[j].q[0] = *(const uint4*)(bp);
      b[j].q[1] = *(const uint4*)(bp + 16);
      __builtin_prefetch(bp + 256, 0, 0);   // -> global_prefetch_b8
    }
    // 8 WMMAs reuse fragments 2x (A) and 4x (B)
#pragma unroll
    for (int i = 0; i < MT; ++i)
#pragma unroll
      for (int j = 0; j < NT; ++j)
        acc[i][j] = __builtin_amdgcn_wmma_f32_16x16x32_bf16(
            false, a[i].v, false, b[j].v, (short)0, acc[i][j], false, false);
  }

  // ---- epilogue: guarded stores ----
#pragma unroll
  for (int j = 0; j < NT; ++j) {
    int n = col0 + (j << 4) + mi;
    if (n >= N) continue;
    float bs = bias ? bias[n] : 0.0f;
#pragma unroll
    for (int i = 0; i < MT; ++i) {
      int mBase = row0 + (i << 4);
      if (mBase >= M) continue;
#pragma unroll
      for (int r = 0; r < 8; ++r) {        // C/D: VGPR r -> M = r + kg*8
        int m = mBase + r + (kg << 3);
        float v = acc[i][j][r] + bs;
        if (C)   C[(long long)m * ldc + n] = v;
        if (Cbf) Cbf[(long long)m * ldcb + n] = (bf16)v;
      }
    }
  }
}

// ---- edge pass 1: e = leaky(a_src[s]+a_dst[d]); segment max into keys ------
__global__ void k_edge_max(const int* __restrict__ ei, int E,
                           const float* __restrict__ as,
                           const float* __restrict__ ad,
                           unsigned* __restrict__ keys) {
  int t = blockIdx.x * blockDim.x + threadIdx.x;
  if (t >= E * HC) return;
  int e = t >> 8, c = t & (HC - 1);
  int s = ei[e], d = ei[E + e];
  float v = as[(long long)s * HC + c] + ad[(long long)d * HC + c];
  v = (v >= 0.f) ? v : NEG_SLOPE * v;
  atomicMax(&keys[(long long)d * HC + c], fkey(v));
}

// ---- edge pass 2: p = exp(e - emax[d]); denom += p; agg += xh[s]*p ---------
__global__ void k_edge_accum(const int* __restrict__ ei, int E,
                             const float* __restrict__ as,
                             const float* __restrict__ ad,
                             const unsigned* __restrict__ keys,
                             const float* __restrict__ xh,
                             float* __restrict__ denom,
                             float* __restrict__ agg) {
  int t = blockIdx.x * blockDim.x + threadIdx.x;
  if (t >= E * HC) return;
  int e = t >> 8, c = t & (HC - 1);
  int s = ei[e], d = ei[E + e];
  long long di = (long long)d * HC + c;
  float v = as[(long long)s * HC + c] + ad[di];
  v = (v >= 0.f) ? v : NEG_SLOPE * v;
  unsigned k = keys[di];
  float mx = k ? fdecode(k) : 0.f;      // untouched segment -> emax = 0
  float p = __expf(v - mx);
  atomicAdd(&denom[di], p);
  atomicAdd(&agg[di], xh[(long long)s * HC + c] * p);
}

// ---- out = agg / (denom + eps) + bias, in place ----------------------------
__global__ void k_finalize(float* __restrict__ agg,
                           const float* __restrict__ denom,
                           const float* __restrict__ bias, int total) {
  int t = blockIdx.x * blockDim.x + threadIdx.x;
  if (t >= total) return;
  agg[t] = agg[t] / (denom[t] + 1e-16f) + bias[t & (HC - 1)];
}

// ---- gather B selected rows into f32 and/or bf16 (with column offset) ------
__global__ void k_select(const float* __restrict__ nodeOut,
                         const int* __restrict__ center,
                         float* __restrict__ dstF,
                         bf16* __restrict__ dstBf, int ldbf, int colOff, int B) {
  int t = blockIdx.x * blockDim.x + threadIdx.x;
  if (t >= B * HC) return;
  int b = t >> 8, c = t & (HC - 1);
  float v = nodeOut[(long long)center[b] * HC + c];
  if (dstF)  dstF[t] = v;
  if (dstBf) dstBf[(long long)b * ldbf + colOff + c] = (bf16)v;
}

// ---- build concat inputs: cat1 = [stp_1 | stp_rw], cat2 left = pp_x --------
__global__ void k_build_cats(const float* __restrict__ outs,
                             bf16* __restrict__ cat1,
                             bf16* __restrict__ cat2, int B) {
  int t = blockIdx.x * blockDim.x + threadIdx.x;
  if (t >= B * HC) return;
  int b = t >> 8, c = t & (HC - 1);
  const long long S = (long long)B * HC;
  float s1 = (outs[t] + outs[S + t] + outs[2 * S + t]) * (1.f / 3.f);
  float s2 = (outs[3 * S + t] + outs[4 * S + t] + outs[5 * S + t]) * (1.f / 3.f);
  cat1[(long long)b * 512 + c]        = (bf16)s1;
  cat1[(long long)b * 512 + 256 + c]  = (bf16)s2;
  cat2[(long long)b * 512 + c]        = (bf16)outs[6 * S + t];
}

// ---------------------------------------------------------------------------
extern "C" void kernel_launch(void* const* d_in, const int* in_sizes, int n_in,
                              void* d_out, int out_size, void* d_ws, size_t ws_size,
                              hipStream_t stream) {
  const int*   g_node_idx   = (const int*)d_in[0];
  const int*   g_edge_index = (const int*)d_in[1];
  const int*   g_center     = (const int*)d_in[2];
  const int*   u_eindex     = (const int*)d_in[3];
  const int*   anchors      = (const int*)d_in[4];
  const float* item_emb     = (const float*)d_in[5];
  const float* user_emb     = (const float*)d_in[6];
  const float* convs_W      = (const float*)d_in[7];
  const float* convs_att_src= (const float*)d_in[8];
  const float* convs_att_dst= (const float*)d_in[9];
  const float* convs_bias   = (const float*)d_in[10];
  const float* u_W          = (const float*)d_in[11];
  const float* u_att_src    = (const float*)d_in[12];
  const float* u_att_dst    = (const float*)d_in[13];
  const float* u_bias       = (const float*)d_in[14];
  const float* a_rw_W       = (const float*)d_in[15];
  const float* a_rw_b       = (const float*)d_in[16];
  const float* pp_stp_W     = (const float*)d_in[17];
  const float* pp_stp_b     = (const float*)d_in[18];
  const float* final_W      = (const float*)d_in[19];
  const float* final_b      = (const float*)d_in[20];

  const int N     = in_sizes[0] / 7;            // 20000
  const int E     = in_sizes[1] / 14;           // 100000
  const int B     = in_sizes[2] / 7;            // 1024
  const int Eu    = in_sizes[3] / 2;            // 100000
  const int USERS = in_sizes[6] / FIN;          // 10000
  const int ITEMS = in_sizes[19] / 512;         // 50000

  // ---- workspace carve ----
  char* wp = (char*)d_ws;
  auto carve = [&](size_t bytes) -> void* {
    void* r = (void*)wp;
    wp += (bytes + 255) & ~(size_t)255;
    return r;
  };
  bf16* wt_convs  = (bf16*)carve((size_t)7 * 256 * 128 * sizeof(bf16));
  bf16* at_src_t  = (bf16*)carve((size_t)7 * HEADS * CH * CH * sizeof(bf16));
  bf16* at_dst_t  = (bf16*)carve((size_t)7 * HEADS * CH * CH * sizeof(bf16));
  bf16* wt_u      = (bf16*)carve((size_t)256 * 128 * sizeof(bf16));
  bf16* ua_src_t  = (bf16*)carve((size_t)HEADS * CH * CH * sizeof(bf16));
  bf16* ua_dst_t  = (bf16*)carve((size_t)HEADS * CH * CH * sizeof(bf16));
  bf16* wt_arw    = (bf16*)carve((size_t)256 * 512 * sizeof(bf16));
  bf16* wt_pp     = (bf16*)carve((size_t)256 * 512 * sizeof(bf16));
  bf16* wt_final  = (bf16*)carve((size_t)ITEMS * 512 * sizeof(bf16));
  bf16* x_bf      = (bf16*)carve((size_t)N * FIN * sizeof(bf16));
  float* xh_f32   = (float*)carve((size_t)N * HC * sizeof(float));
  bf16* xh_bf     = (bf16*)carve((size_t)N * HC * sizeof(bf16));
  float* a_src    = (float*)carve((size_t)N * HC * sizeof(float));
  float* a_dst    = (float*)carve((size_t)N * HC * sizeof(float));
  unsigned* keys  = (unsigned*)carve((size_t)N * HC * sizeof(unsigned));
  float* denom    = (float*)carve((size_t)N * HC * sizeof(float));
  float* agg      = (float*)carve((size_t)N * HC * sizeof(float));
  float* outs     = (float*)carve((size_t)7 * B * HC * sizeof(float));
  bf16* cat1      = (bf16*)carve((size_t)B * 512 * sizeof(bf16));
  bf16* cat2      = (bf16*)carve((size_t)B * 512 * sizeof(bf16));
  bf16* frepr     = (bf16*)carve((size_t)B * 512 * sizeof(bf16));

  const int TPB = 256;
  auto g1 = [](long long n, int tpb) { return dim3((unsigned)((n + tpb - 1) / tpb)); };
  auto cdiv = [](int a, int b) { return (a + b - 1) / b; };

  // ---- one-time (per call) weight convert+transpose into bf16 ----
  k_transpose_bf16<<<dim3(g1(128 * 256, TPB).x, 1, 7), TPB, 0, stream>>>(convs_W, wt_convs, 128, 256);
  k_transpose_bf16<<<dim3(g1(CH * CH, TPB).x, 1, 28), TPB, 0, stream>>>(convs_att_src, at_src_t, CH, CH);
  k_transpose_bf16<<<dim3(g1(CH * CH, TPB).x, 1, 28), TPB, 0, stream>>>(convs_att_dst, at_dst_t, CH, CH);
  k_transpose_bf16<<<dim3(g1(128 * 256, TPB).x, 1, 1), TPB, 0, stream>>>(u_W, wt_u, 128, 256);
  k_transpose_bf16<<<dim3(g1(CH * CH, TPB).x, 1, 4), TPB, 0, stream>>>(u_att_src, ua_src_t, CH, CH);
  k_transpose_bf16<<<dim3(g1(CH * CH, TPB).x, 1, 4), TPB, 0, stream>>>(u_att_dst, ua_dst_t, CH, CH);
  k_transpose_bf16<<<dim3(g1(512 * 256, TPB).x, 1, 1), TPB, 0, stream>>>(a_rw_W, wt_arw, 512, 256);
  k_transpose_bf16<<<dim3(g1(512 * 256, TPB).x, 1, 1), TPB, 0, stream>>>(pp_stp_W, wt_pp, 512, 256);
  k_transpose_bf16<<<dim3(g1((long long)512 * ITEMS, TPB).x, 1, 1), TPB, 0, stream>>>(final_W, wt_final, 512, ITEMS);

  // GEMM launcher: grid (M/64, N/128, z), dynamic LDS = 64*(K+pad)*2 bytes
  auto gemm = [&](const bf16* A, int lda, const bf16* Bt, int ldbt,
                  float* C, int ldc, bf16* Cbf, int ldcb, const float* bias,
                  int M, int Nc, int K, int zdim, int zA, int zB, int zC) {
    dim3 grid(cdiv(M, BLK_ROWS), cdiv(Nc, BLK_COLS), zdim);
    size_t shmem = (size_t)BLK_ROWS * (K + LDS_PAD) * sizeof(bf16);
    k_wmma_gemm<<<grid, 128, shmem, stream>>>(A, lda, Bt, ldbt, C, ldc,
                                              Cbf, ldcb, bias, M, Nc, K,
                                              zA, zB, zC);
  };

  // ---- one DGAT conv: shared routine over (emb, idx, weights, edges) ----
  auto run_dgat = [&](const float* emb, const int* idx, int nNodes,
                      const bf16* Wt, const bf16* As_t, const bf16* Ad_t,
                      const float* bias, const int* eidx, int nE) {
    // gather + convert x
    k_gather_bf16<<<g1((long long)nNodes * FIN, TPB), TPB, 0, stream>>>(emb, idx, x_bf, nNodes, FIN);
    // xh = x @ W  (emit f32 + bf16)
    gemm(x_bf, FIN, Wt, FIN, xh_f32, HC, xh_bf, HC, nullptr,
         nNodes, HC, FIN, 1, 0, 0, 0);
    // per-head attention GEMMs: a_src / a_dst  (z = head)
    gemm(xh_bf, HC, As_t, CH, a_src, HC, nullptr, 0, nullptr,
         nNodes, CH, CH, HEADS, CH, CH * CH, CH);
    gemm(xh_bf, HC, Ad_t, CH, a_dst, HC, nullptr, 0, nullptr,
         nNodes, CH, CH, HEADS, CH, CH * CH, CH);
    // segment softmax + aggregation
    (void)hipMemsetAsync(keys, 0, (size_t)nNodes * HC * sizeof(unsigned), stream);
    (void)hipMemsetAsync(denom, 0, (size_t)nNodes * HC * sizeof(float), stream);
    (void)hipMemsetAsync(agg, 0, (size_t)nNodes * HC * sizeof(float), stream);
    k_edge_max<<<g1((long long)nE * HC, TPB), TPB, 0, stream>>>(eidx, nE, a_src, a_dst, keys);
    k_edge_accum<<<g1((long long)nE * HC, TPB), TPB, 0, stream>>>(eidx, nE, a_src, a_dst, keys, xh_f32, denom, agg);
    k_finalize<<<g1((long long)nNodes * HC, TPB), TPB, 0, stream>>>(agg, denom, bias, nNodes * HC);
  };

  // ---- 7 item-graph convolutions ----
  for (int g = 0; g < 7; ++g) {
    run_dgat(item_emb, g_node_idx + (size_t)g * N, N,
             wt_convs + (size_t)g * 256 * 128,
             at_src_t + (size_t)g * HEADS * CH * CH,
             at_dst_t + (size_t)g * HEADS * CH * CH,
             convs_bias + (size_t)g * HC,
             g_edge_index + (size_t)g * 2 * E, E);
    k_select<<<g1((long long)B * HC, TPB), TPB, 0, stream>>>(
        agg, g_center + (size_t)g * B, outs + (size_t)g * B * HC, nullptr, 0, 0, B);
  }

  // ---- user convolution -> u_x straight into final_repr right half (bf16) --
  run_dgat(user_emb, nullptr, USERS, wt_u, ua_src_t, ua_dst_t, u_bias, u_eindex, Eu);
  k_select<<<g1((long long)B * HC, TPB), TPB, 0, stream>>>(
      agg, anchors, nullptr, frepr, 512, 256, B);

  // ---- MLP head ----
  k_build_cats<<<g1((long long)B * HC, TPB), TPB, 0, stream>>>(outs, cat1, cat2, B);

  // stp = cat1 @ a_rw_W + b   -> bf16 into cat2 right half
  gemm(cat1, 512, wt_arw, 512, nullptr, 0, cat2 + 256, 512, a_rw_b,
       B, HC, 512, 1, 0, 0, 0);
  // y = cat2 @ pp_stp_W + b   -> bf16 into final_repr left half
  gemm(cat2, 512, wt_pp, 512, nullptr, 0, frepr, 512, pp_stp_b,
       B, HC, 512, 1, 0, 0, 0);
  // logits = final_repr @ final_W + final_b  -> d_out [B, ITEMS] f32
  gemm(frepr, 512, wt_final, 512, (float*)d_out, ITEMS, nullptr, 0, final_b,
       B, ITEMS, 512, 1, 0, 0, 0);
}
